// MoLAdapter_3083786519120
// MI455X (gfx1250) — matrix operand order repaired and checked by reading, compiled, not run on hardware
//
#include <hip/hip_runtime.h>
#include <hip/hip_bf16.h>
#include <math.h>

// ---------- problem constants (B=4,S=4096,D=2048,E=8,R=16,K=2) ----------
#define NTOK   16384      // B*S tokens
#define DIM    2048       // hidden
#define NEXP   8
#define RANK   16
#define ERK    128        // E*R (down-proj columns / up-proj K)
#define NCOMB  144        // ERK + 8 router cols + 8 zero pad (tile aligned)

typedef __attribute__((ext_vector_type(16))) __bf16 v16bf;
typedef __attribute__((ext_vector_type(8)))  float  v8f;
typedef __attribute__((ext_vector_type(4)))  float  v4f;

// workspace layout (bytes)
#define WS_WCOMBT 0                         // NCOMB*DIM bf16      = 589824
#define WS_UPT    589824                    // DIM*ERK  bf16       = 524288
#define WS_H      1114112                   // NTOK*ERK f32        = 8388608
#define WS_LOGITS 9502720                   // NTOK*NEXP f32       = 524288
#define WS_WFULL  10027008                  // NTOK*NEXP f32       = 524288

// ---------------------------------------------------------------------------
// Kernel 0: build bf16 transposed weight images.
//   WcombT[n][k]: n<128 -> experts_down[e][k][r] (e=n/16, r=n%16)
//                 128<=n<136 -> router_w[n-128][k];  n>=136 -> 0
//   upT[d][n]   : experts_up[e][r][d]
// ---------------------------------------------------------------------------
__global__ __launch_bounds__(256) void prep_weights(
    const float* __restrict__ router_w, const float* __restrict__ down,
    const float* __restrict__ up, __bf16* __restrict__ wcombT,
    __bf16* __restrict__ upT) {
  int idx = blockIdx.x * 256 + threadIdx.x;
  const int total1 = NCOMB * DIM;
  if (idx < total1) {
    int n = idx / DIM, k = idx % DIM;
    float v;
    if (n < ERK) {
      int e = n >> 4, r = n & 15;
      v = down[(size_t)e * DIM * RANK + (size_t)k * RANK + r];
    } else if (n < ERK + NEXP) {
      v = router_w[(size_t)(n - ERK) * DIM + k];
    } else {
      v = 0.0f;
    }
    wcombT[idx] = (__bf16)v;
  }
  int idx2 = idx - total1;
  const int total2 = DIM * ERK;
  if (idx2 >= 0 && idx2 < total2) {
    int d = idx2 / ERK, n = idx2 % ERK;
    int e = n >> 4, r = n & 15;
    upT[idx2] = (__bf16)up[(size_t)e * RANK * DIM + (size_t)r * DIM + d];
  }
}

// ---------------------------------------------------------------------------
// Kernel 1: fused down-proj + router GEMM.
//   [NTOK x DIM] (bf16 from f32 x) @ WcombT^T -> h[NTOK x 128], logits[NTOK x 8]
//   grid 256 blocks x 128 thr (4 waves). Wave owns 16 tokens x 144 cols.
//   x stream prefetched 16 k-iters (2 KB/row) ahead (global_prefetch_b8);
//   k-loop unrolled 2x so two chunks of loads/WMMAs stay in flight.
// ---------------------------------------------------------------------------
__global__ __launch_bounds__(128) void down_router(
    const float* __restrict__ x, const __bf16* __restrict__ wcombT,
    float* __restrict__ h, float* __restrict__ logits) {
  const int wave  = threadIdx.x >> 5;
  const int lane  = threadIdx.x & 31;
  const int row   = lane & 15;          // matrix row within 16-token tile
  const int khalf = lane >> 4;          // 0 or 1 (lane-half K split)
  const int mbase = blockIdx.x * 64 + wave * 16;
  const float* xrow = x + (size_t)(mbase + row) * DIM;

  v8f acc[9] = {};                      // 9 N-tiles (8 down + 1 router/pad)

#pragma unroll 2
  for (int k0 = 0; k0 < DIM; k0 += 32) {
    // prefetch x 512 floats ahead (clamped in-bounds; line = 32 floats)
    {
      int kpf = k0 + 512;
      if (kpf >= DIM) kpf = DIM - 32;
      __builtin_prefetch(xrow + kpf, 0, 0);
    }
    // A fragment: lane(row) holds K = khalf*16+{0..15}
    const int ka = k0 + khalf * 16;
    v4f x0 = *(const v4f*)(xrow + ka);
    v4f x1 = *(const v4f*)(xrow + ka + 4);
    v4f x2 = *(const v4f*)(xrow + ka + 8);
    v4f x3 = *(const v4f*)(xrow + ka + 12);
    v16bf a;
#pragma unroll
    for (int i = 0; i < 4; ++i) {
      a[i]      = (__bf16)x0[i];
      a[4 + i]  = (__bf16)x1[i];
      a[8 + i]  = (__bf16)x2[i];
      a[12 + i] = (__bf16)x3[i];
    }
    const int kb = k0 + khalf * 16;     // B: lane-half holds 16 contiguous K
#pragma unroll
    for (int nt = 0; nt < 9; ++nt) {
      const int col = nt * 16 + row;
      v16bf b = *(const v16bf*)(wcombT + (size_t)col * DIM + kb); // 32B aligned
      acc[nt] = __builtin_amdgcn_wmma_f32_16x16x32_bf16(
          false, a, false, b, (short)0, acc[nt], false, false);
    }
  }

  // C/D layout: lane holds col = row, VGPR v -> matrix row v + 8*khalf
#pragma unroll
  for (int nt = 0; nt < 8; ++nt) {
    const int col = nt * 16 + row;
#pragma unroll
    for (int v = 0; v < 8; ++v)
      h[(size_t)(mbase + v + 8 * khalf) * ERK + col] = acc[nt][v];
  }
  if (row < NEXP) {                      // router logits tile (cols 128..135)
#pragma unroll
    for (int v = 0; v < 8; ++v)
      logits[(size_t)(mbase + v + 8 * khalf) * NEXP + row] = acc[8][v];
  }
}

// ---------------------------------------------------------------------------
// Kernel 2: per-token top-2 + softmax -> dense routing weights (zeros else).
// ---------------------------------------------------------------------------
__global__ __launch_bounds__(256) void topk_softmax(
    const float* __restrict__ logits, float* __restrict__ wfull) {
  int t = blockIdx.x * 256 + threadIdx.x;
  if (t >= NTOK) return;
  float l[NEXP];
#pragma unroll
  for (int i = 0; i < NEXP; ++i) l[i] = logits[(size_t)t * NEXP + i];
  int i0 = 0;
#pragma unroll
  for (int i = 1; i < NEXP; ++i) if (l[i] > l[i0]) i0 = i;   // first max (ties)
  int i1 = (i0 == 0) ? 1 : 0;
#pragma unroll
  for (int i = 0; i < NEXP; ++i)
    if (i != i0 && l[i] > l[i1]) i1 = i;
  float e1 = expf(l[i1] - l[i0]);       // e0 = 1
  float inv = 1.0f / (1.0f + e1);
  float o[NEXP];
#pragma unroll
  for (int i = 0; i < NEXP; ++i) o[i] = 0.0f;
  o[i0] = inv;
  o[i1] = e1 * inv;
#pragma unroll
  for (int i = 0; i < NEXP; ++i) wfull[(size_t)t * NEXP + i] = o[i];
}

// ---------------------------------------------------------------------------
// Kernel 3: up-proj + residual.  out = x + (h .* w_full) @ up  (K = 128)
//   grid 1024 blocks (16-token tiles) x 256 thr (8 waves); wave owns 256 cols.
//   Routing weight fused at A-fragment build: expert = K-index / 16.
//   Cache policy: x read non-temporal (last use -- keep L2 for streams),
//   out written non-temporal (never re-read -- don't evict L2-resident data).
//   nt-loop unrolled 2x for two independent B/x/out streams in flight.
// ---------------------------------------------------------------------------
__global__ __launch_bounds__(256) void up_residual(
    const float* __restrict__ x, const float* __restrict__ h,
    const float* __restrict__ wfull, const __bf16* __restrict__ upT,
    float* __restrict__ out) {
  const int wave  = threadIdx.x >> 5;
  const int lane  = threadIdx.x & 31;
  const int row   = lane & 15;
  const int khalf = lane >> 4;
  const int mbase = blockIdx.x * 16;
  const int t     = mbase + row;
  const float* hrow = h + (size_t)t * ERK;
  const float* wrow = wfull + (size_t)t * NEXP;

  // 4 A fragments covering K = 0..127 (g = h * w_e), built once per wave
  v16bf a[4];
#pragma unroll
  for (int ks = 0; ks < 4; ++ks) {
    const int k0 = ks * 32;
    const int ka = k0 + khalf * 16;
    const int e0 = k0 >> 4;             // expert for K in [k0, k0+16)
    const float w0 = wrow[e0 + khalf];
    const float w1 = w0;                // lane half covers one expert's 16 K
    v4f h0 = *(const v4f*)(hrow + ka);
    v4f h1 = *(const v4f*)(hrow + ka + 4);
    v4f h2 = *(const v4f*)(hrow + ka + 8);
    v4f h3 = *(const v4f*)(hrow + ka + 12);
#pragma unroll
    for (int i = 0; i < 4; ++i) {
      a[ks][i]      = (__bf16)(h0[i] * w0);
      a[ks][4 + i]  = (__bf16)(h1[i] * w0);
      a[ks][8 + i]  = (__bf16)(h2[i] * w1);
      a[ks][12 + i] = (__bf16)(h3[i] * w1);
    }
  }

  const int ncolbase = wave * 256;
#pragma unroll 2
  for (int nt = 0; nt < 16; ++nt) {
    const int col = ncolbase + nt * 16 + row;
    v8f c = {};
#pragma unroll
    for (int ks = 0; ks < 4; ++ks) {
      const int kb = ks * 32 + khalf * 16;
      v16bf b = *(const v16bf*)(upT + (size_t)col * ERK + kb); // 32B aligned
      c = __builtin_amdgcn_wmma_f32_16x16x32_bf16(
          false, a[ks], false, b, (short)0, c, false, false);
    }
#pragma unroll
    for (int v = 0; v < 8; ++v) {
      const size_t o = (size_t)(mbase + v + 8 * khalf) * DIM + col;
      float xi = __builtin_nontemporal_load(x + o);   // last use of x
      __builtin_nontemporal_store(xi + c[v], out + o); // streaming store
    }
  }
}

// ---------------------------------------------------------------------------
extern "C" void kernel_launch(void* const* d_in, const int* in_sizes, int n_in,
                              void* d_out, int out_size, void* d_ws, size_t ws_size,
                              hipStream_t stream) {
  (void)in_sizes; (void)n_in; (void)out_size; (void)ws_size;
  const float* x        = (const float*)d_in[0];  // (B,S,D)
  const float* router_w = (const float*)d_in[1];  // (E,D)
  const float* down     = (const float*)d_in[2];  // (E,D,R)
  const float* up       = (const float*)d_in[3];  // (E,R,D)
  float* out = (float*)d_out;

  char* ws = (char*)d_ws;
  __bf16* wcombT = (__bf16*)(ws + WS_WCOMBT);
  __bf16* upT    = (__bf16*)(ws + WS_UPT);
  float*  h      = (float*)(ws + WS_H);
  float*  logits = (float*)(ws + WS_LOGITS);
  float*  wfull  = (float*)(ws + WS_WFULL);

  // 0) bf16 transposed weight images (294912 + 262144 elems)
  prep_weights<<<2176, 256, 0, stream>>>(router_w, down, up, wcombT, upT);
  // 1) fused down-proj + router GEMM (reads x once, stays L2-resident)
  down_router<<<NTOK / 64, 128, 0, stream>>>(x, wcombT, h, logits);
  // 2) top-2 softmax routing weights
  topk_softmax<<<NTOK / 256, 256, 0, stream>>>(logits, wfull);
  // 3) up-proj + residual (NT re-read of x, NT write of out)
  up_residual<<<NTOK / 16, 256, 0, stream>>>(x, h, wfull, upT, out);
}
